// ScaledDotProductAttention_63660005261368
// MI455X (gfx1250) — matrix operand order repaired and checked by reading, not compiled, not run on hardware
//
#include <hip/hip_runtime.h>

typedef __attribute__((ext_vector_type(16))) _Float16 v16h;
typedef __attribute__((ext_vector_type(8)))  float    v8f;

#define SDIM 2048
#define DDIM 128
#define BM   128   // query rows per workgroup (16 per wave x 8 waves)
#define BN   64    // key block: 4 score tiles, two K=32 chunks for PV

struct alignas(16) U4  { unsigned int a, b, c, d; };
struct U8x { U4 lo, hi; };   // 32 bytes == v16h

// Assemble a v16h WMMA operand from two 16-byte LDS chunks (two ds_load_b128).
__device__ inline v16h ld_op(const _Float16* p0, const _Float16* p1) {
    U8x u;
    u.lo = *(const U4*)p0;
    u.hi = *(const U4*)p1;
    return __builtin_bit_cast(v16h, u);
}

__global__ void __launch_bounds__(256)
fattn_f16wmma_kernel(const float* __restrict__ Q, const float* __restrict__ K,
                     const float* __restrict__ V, float* __restrict__ O)
{
    __shared__ _Float16 Ks[2][BN * DDIM];   // 2 x 16 KB : K block [key][d], double-buffered
    __shared__ _Float16 Vt[2][DDIM * BN];   // 2 x 16 KB : V block transposed [d][key]
    __shared__ _Float16 Ps[8][16 * BN];     // 16 KB     : per-wave probs (C->A relayout)

    const int tid   = threadIdx.x;
    const int lane  = tid & 31;
    const int wv    = tid >> 5;      // wave id 0..7
    const int l15   = lane & 15;
    const int hsel  = lane >> 4;     // half-wave select (0/1)
    const int bh    = blockIdx.y;    // fused batch*head
    const int qBase = blockIdx.x * BM;

    const float scale = 0.08838834764831845f; // 1/sqrt(128)

    const float* Qg = Q + (size_t)bh * SDIM * DDIM;
    const float* Kg = K + (size_t)bh * SDIM * DDIM;
    const float* Vg = V + (size_t)bh * SDIM * DDIM;
    float*       Og = O + (size_t)bh * SDIM * DDIM;

    const int q0w = qBase + wv * 16;        // this wave's first query row
    const int nkb = (qBase + BM) / BN;      // causal: keys [0, qBase+128)

    // ---- Q tile: global fp32 -> f16 A-matrix layout, in registers (wave-private) ----
    v16h qa[4];
    {
        const float* qrow = Qg + (size_t)(q0w + l15) * DDIM + 8 * hsel;
        #pragma unroll
        for (int c = 0; c < 4; ++c) {
            const float* p = qrow + 32 * c;
            float4 x0 = *(const float4*)(p);        // K = 8h .. 8h+3
            float4 x1 = *(const float4*)(p + 4);    // K = 8h+4 .. 8h+7
            float4 x2 = *(const float4*)(p + 16);   // K = 8h+16 .. 8h+19
            float4 x3 = *(const float4*)(p + 20);   // K = 8h+20 .. 8h+23
            v16h a;
            a[0]  = (_Float16)(x0.x * scale); a[1]  = (_Float16)(x0.y * scale);
            a[2]  = (_Float16)(x0.z * scale); a[3]  = (_Float16)(x0.w * scale);
            a[4]  = (_Float16)(x1.x * scale); a[5]  = (_Float16)(x1.y * scale);
            a[6]  = (_Float16)(x1.z * scale); a[7]  = (_Float16)(x1.w * scale);
            a[8]  = (_Float16)(x2.x * scale); a[9]  = (_Float16)(x2.y * scale);
            a[10] = (_Float16)(x2.z * scale); a[11] = (_Float16)(x2.w * scale);
            a[12] = (_Float16)(x3.x * scale); a[13] = (_Float16)(x3.y * scale);
            a[14] = (_Float16)(x3.z * scale); a[15] = (_Float16)(x3.w * scale);
            qa[c] = a;
        }
    }

    // ---- stage one 64-key K/V block into LDS buffer `buf` ----
    auto stage = [&](int b, int buf) {
        #pragma unroll
        for (int i = 0; i < 8; ++i) {
            int idx = tid + 256 * i;        // 2048 float4 per block
            int row = idx >> 5;             // key-local 0..63
            int c4  = (idx & 31) * 4;
            size_t goff = (size_t)(b * BN + row) * DDIM + c4;
            float4 k4 = *(const float4*)(Kg + goff);
            _Float16* kd = &Ks[buf][row * DDIM + c4];
            kd[0] = (_Float16)k4.x; kd[1] = (_Float16)k4.y;
            kd[2] = (_Float16)k4.z; kd[3] = (_Float16)k4.w;
            float4 v4 = *(const float4*)(Vg + goff);
            Vt[buf][(c4 + 0) * BN + row] = (_Float16)v4.x;
            Vt[buf][(c4 + 1) * BN + row] = (_Float16)v4.y;
            Vt[buf][(c4 + 2) * BN + row] = (_Float16)v4.z;
            Vt[buf][(c4 + 3) * BN + row] = (_Float16)v4.w;
        }
        if (b + 1 < nkb) {                  // one 128B line per thread over next block
            size_t nb = (size_t)(b + 1) * BN * DDIM + (size_t)tid * 32;
            __builtin_prefetch(Kg + nb, 0, 1);
            __builtin_prefetch(Vg + nb, 0, 1);
        }
    };

    const v8f vzero = {0.f, 0.f, 0.f, 0.f, 0.f, 0.f, 0.f, 0.f};
    float m[8], l[8];
    v8f acc[8];                             // 8 D-column tiles of 16x16 f32
    #pragma unroll
    for (int v = 0; v < 8; ++v) { m[v] = -3.0e38f; l[v] = 0.f; }
    #pragma unroll
    for (int d = 0; d < 8; ++d) acc[d] = vzero;

    stage(0, 0);
    __syncthreads();

    for (int kb = 0; kb < nkb; ++kb) {
        const int cur   = kb & 1;
        const int kbase = kb * BN;

        // overlap: stage next block into the idle buffer (inactive waves only stage)
        if (kb + 1 < nkb) stage(kb + 1, cur ^ 1);

        if (kbase <= q0w + 15) {            // wave-level causal early-out
            // ---- S = (Q*scale) K^T : four 16x16 score tiles over K=128 ----
            v8f s[4];
            #pragma unroll
            for (int t = 0; t < 4; ++t) s[t] = vzero;
            #pragma unroll
            for (int c = 0; c < 4; ++c) {
                #pragma unroll
                for (int t = 0; t < 4; ++t) {
                    const _Float16* bp = &Ks[cur][(16 * t + l15) * DDIM + 32 * c + 8 * hsel];
                    v16h b = ld_op(bp, bp + 16);
                    s[t] = __builtin_amdgcn_wmma_f32_16x16x32_f16(false, qa[c], false, b,
                                                                  (short)0, s[t], false, false);
                }
            }

            // ---- causal mask: only diagonal blocks pay the cmp/cndmask cost ----
            const bool partial = (kbase + BN - 1) > q0w;   // wave-uniform
            if (partial) {
                #pragma unroll
                for (int t = 0; t < 4; ++t) {
                    int colt = kbase + 16 * t + l15;
                    #pragma unroll
                    for (int v = 0; v < 8; ++v) {
                        int rowq = q0w + v + 8 * hsel;
                        if (colt > rowq) s[t][v] = -1.0e30f;
                    }
                }
            }

            // ---- online softmax (16-lane row reductions) ----
            float rmax[8], alpha[8], rsum[8];
            #pragma unroll
            for (int v = 0; v < 8; ++v)
                rmax[v] = fmaxf(fmaxf(s[0][v], s[1][v]), fmaxf(s[2][v], s[3][v]));
            #pragma unroll
            for (int off = 8; off >= 1; off >>= 1)
                #pragma unroll
                for (int v = 0; v < 8; ++v)
                    rmax[v] = fmaxf(rmax[v], __shfl_xor(rmax[v], off, 16));
            #pragma unroll
            for (int v = 0; v < 8; ++v) {
                float mn = fmaxf(m[v], rmax[v]);
                alpha[v] = __expf(m[v] - mn);
                m[v] = mn;
                float p0 = __expf(s[0][v] - mn);
                float p1 = __expf(s[1][v] - mn);
                float p2 = __expf(s[2][v] - mn);
                float p3 = __expf(s[3][v] - mn);
                s[0][v] = p0; s[1][v] = p1; s[2][v] = p2; s[3][v] = p3;
                rsum[v] = (p0 + p1) + (p2 + p3);
            }
            #pragma unroll
            for (int off = 8; off >= 1; off >>= 1)
                #pragma unroll
                for (int v = 0; v < 8; ++v)
                    rsum[v] += __shfl_xor(rsum[v], off, 16);
            #pragma unroll
            for (int v = 0; v < 8; ++v) l[v] = l[v] * alpha[v] + rsum[v];
            #pragma unroll
            for (int d = 0; d < 8; ++d)
                #pragma unroll
                for (int v = 0; v < 8; ++v)
                    acc[d][v] *= alpha[v];

            // ---- P: C-layout -> f16 A-layout via wave-private LDS scratch ----
            _Float16* Pw = Ps[wv];
            #pragma unroll
            for (int t = 0; t < 4; ++t)
                #pragma unroll
                for (int v = 0; v < 8; ++v)
                    Pw[(v + 8 * hsel) * BN + 16 * t + l15] = (_Float16)s[t][v];
            // same-wave DS ordering: compiler inserts s_wait_dscnt
            const _Float16* pap = &Pw[l15 * BN + 8 * hsel];
            v16h pa0 = ld_op(pap,      pap + 16);   // keys 0..31 of block
            v16h pa1 = ld_op(pap + 32, pap + 48);   // keys 32..63 of block

            // ---- acc += P V  (8 D-column tiles x two K=32 chunks) ----
            #pragma unroll
            for (int d = 0; d < 8; ++d) {
                const _Float16* vb = &Vt[cur][(d * 16 + l15) * BN + 8 * hsel];
                v16h b0 = ld_op(vb,      vb + 16);
                acc[d] = __builtin_amdgcn_wmma_f32_16x16x32_f16(false, pa0, false, b0,
                                                                (short)0, acc[d], false, false);
                v16h b1 = ld_op(vb + 32, vb + 48);
                acc[d] = __builtin_amdgcn_wmma_f32_16x16x32_f16(false, pa1, false, b1,
                                                                (short)0, acc[d], false, false);
            }
        }
        __syncthreads();   // staging of next buffer done; current buffer free for reuse
    }

    // ---- epilogue: normalize and store fp32 output ----
    #pragma unroll
    for (int v = 0; v < 8; ++v) {
        float inv = 1.0f / l[v];
        int row = q0w + v + 8 * hsel;
        float* orow = Og + (size_t)row * DDIM + l15;
        #pragma unroll
        for (int d = 0; d < 8; ++d)
            orow[d * 16] = acc[d][v] * inv;
    }
}

extern "C" void kernel_launch(void* const* d_in, const int* in_sizes, int n_in,
                              void* d_out, int out_size, void* d_ws, size_t ws_size,
                              hipStream_t stream) {
    const float* Q = (const float*)d_in[0];
    const float* K = (const float*)d_in[1];
    const float* V = (const float*)d_in[2];
    // d_in[3] is the causal tril mask (bool): computed analytically in-kernel,
    // saving 268 MB of mask traffic.
    float* O = (float*)d_out;
    dim3 grid(SDIM / BM, 4 * 16);  // (q-tiles, B*H) = (16, 64)
    fattn_f16wmma_kernel<<<grid, dim3(256), 0, stream>>>(Q, K, V, O);
}
